// TTLinear_57939108823471
// MI455X (gfx1250) — matrix operand order repaired and checked by reading, not compiled, not run on hardware
//
#include <hip/hip_runtime.h>
#include <stddef.h>

// ---------------------------------------------------------------------------
// TT-linear  y = x @ W^T + bias, W implicitly given by 3 TT-cores.
// Strategy: materialize W (bf16) in workspace, then dense WMMA GEMM with
// async global->LDS copies (double-buffered) on gfx1250.
//   ws layout:  [0,8MB)   Xbf   bf16 [1024][4096]
//               [8,40MB)  Wb    bf16 [4096][4096]   (row o, contiguous i)
//               [40,48MB) A01   f32  [256][256][32] (o01, i01, r2)
// ---------------------------------------------------------------------------

typedef float  v8f   __attribute__((ext_vector_type(8)));
typedef __bf16 v16bf __attribute__((ext_vector_type(16)));
typedef __bf16 v8bf  __attribute__((ext_vector_type(8)));
typedef __bf16 v4bf  __attribute__((ext_vector_type(4)));

union BF16x16 { v16bf v; v8bf h[2]; };

constexpr int BATCH = 1024;
constexpr int INF   = 4096;
constexpr int OUTF  = 4096;

#if defined(__HIP_DEVICE_COMPILE__) && \
    __has_builtin(__builtin_amdgcn_global_load_async_to_lds_b128)
#define USE_ASYNC_LDS 1
// builtin signature (from clang diagnostics, rounds 2-3):
//   (int __attribute__((vector_size(16))) addrspace(1)*,  // global src
//    int __attribute__((vector_size(16))) addrspace(3)*,  // LDS dst
//    imm int offset, imm int cpol)
typedef int gvec4i __attribute__((vector_size(16)));
#define GAS(p) ((__attribute__((address_space(1))) gvec4i*)(p))
#define LAS(p) ((__attribute__((address_space(3))) gvec4i*)(p))
#if __has_builtin(__builtin_amdgcn_s_wait_asynccnt)
#define WAIT_ASYNC(n) __builtin_amdgcn_s_wait_asynccnt(n)
#else
#define WAIT_ASYNC(n) asm volatile("s_wait_asynccnt %0" ::"i"(n) : "memory")
#endif
#endif

// ---------------- x (f32) -> bf16, vectorized ------------------------------
__global__ void tt_cast_x(const float* __restrict__ x, __bf16* __restrict__ xb) {
    int i = blockIdx.x * blockDim.x + threadIdx.x;            // over n/4
    float4 v = ((const float4*)x)[i];
    v4bf o;
    o.x = (__bf16)v.x; o.y = (__bf16)v.y; o.z = (__bf16)v.z; o.w = (__bf16)v.w;
    ((v4bf*)xb)[i] = o;
}

// ---------------- A01[o01][i01][r2] = sum_r1 core0[o0][i0][r1]*core1[r1][o1][i1][r2]
__global__ void tt_combine01(const float* __restrict__ core0,
                             const float* __restrict__ core1,
                             float* __restrict__ A01) {
    int idx = blockIdx.x * blockDim.x + threadIdx.x;          // 0 .. 2M-1
    int r2  = idx & 31;
    int i01 = (idx >> 5) & 255;
    int o01 = idx >> 13;
    int o0 = o01 >> 4, o1 = o01 & 15;
    int i0 = i01 >> 4, i1 = i01 & 15;
    const float* c0 = core0 + (o0 * 16 + i0) * 32;            // [r1]
    const float* c1 = core1 + (o1 * 16 + i1) * 32 + r2;       // stride r1 = 8192
    float s = 0.f;
#pragma unroll
    for (int r1 = 0; r1 < 32; ++r1)
        s += c0[r1] * c1[(size_t)r1 * 8192];
    A01[idx] = s;
}

// ---------------- W[o][i] = sum_r2 A01[o01][i01][r2] * core2[r2][o2][i2] ----
__global__ void tt_expand_w(const float* __restrict__ A01,
                            const float* __restrict__ core2,
                            __bf16* __restrict__ Wb) {
    int idx = blockIdx.x * blockDim.x + threadIdx.x;          // o*4096 + i
    int i = idx & 4095, o = idx >> 12;
    int o01 = o >> 4, o2 = o & 15;
    int i01 = i >> 4, i2 = i & 15;
    const float* a  = A01 + ((size_t)o01 * 256 + i01) * 32;   // [r2]
    const float* c2 = core2 + o2 * 16 + i2;                   // stride r2 = 256
    float s = 0.f;
#pragma unroll
    for (int r2 = 0; r2 < 32; ++r2)
        s += a[r2] * c2[(size_t)r2 * 256];
    Wb[idx] = (__bf16)s;
}

// ---------------- GEMM:  Y[1024][4096] = Xbf * Wb^T + bias -----------------
// Block tile 128(M) x 128(N), K chunks of 32; 8 waves = 2x4, wave tile 64x32.
// Double-buffered LDS; async global->LDS copies when available.
__global__ __launch_bounds__(256)
void tt_gemm_bias(const __bf16* __restrict__ X,   // [1024][4096]
                  const __bf16* __restrict__ W,   // [4096][4096] row o
                  const float*  __restrict__ bias,
                  float* __restrict__ Y) {
    constexpr int K = INF, N = OUTF;
    constexpr int NC = K / 32;                 // 128 K-chunks
    __shared__ __bf16 sA[2][128][32];          // rows m, contiguous k (16 KB)
    __shared__ __bf16 sB[2][128][32];          // rows n, contiguous k (16 KB)

    const int tid  = threadIdx.x;
    const int lane = tid & 31;
    const int wave = tid >> 5;
    const int wm = wave >> 2, wn = wave & 3;
    const int m0 = blockIdx.y * 128;
    const int n0 = blockIdx.x * 128;

    // cooperative tile copy: each thread moves 32B of A and 32B of B per chunk
    const int lr = tid >> 1;            // row 0..127
    const int lc = (tid & 1) * 16;      // col 0 / 16

    // fragment addressing (ISA 7.12.2, wave32):
    //  A 16x32 bf16: lanes 0-15 -> K {0..7,16..23}; lanes 16-31 -> K {8..15,24..31}
    //  B 32x16 bf16: lanes 0-15 -> K {0..7,16..23}; lanes 16-31 -> K {8..15,24..31}
    const int mrow  = lane & 15;
    const int hi    = lane >> 4;
    const int koffA = hi * 8;
    const int koffB = hi * 8;

    v8f acc[4][2] = {};

    auto issue_copy = [&](int k0, int buf) {
        const __bf16* gA = X + (size_t)(m0 + lr) * K + (k0 + lc);
        const __bf16* gB = W + (size_t)(n0 + lr) * K + (k0 + lc);
#ifdef USE_ASYNC_LDS
        __builtin_amdgcn_global_load_async_to_lds_b128(GAS(gA),     LAS(&sA[buf][lr][lc]),     0, 0);
        __builtin_amdgcn_global_load_async_to_lds_b128(GAS(gA + 8), LAS(&sA[buf][lr][lc + 8]), 0, 0);
        __builtin_amdgcn_global_load_async_to_lds_b128(GAS(gB),     LAS(&sB[buf][lr][lc]),     0, 0);
        __builtin_amdgcn_global_load_async_to_lds_b128(GAS(gB + 8), LAS(&sB[buf][lr][lc + 8]), 0, 0);
#else
        uint4 a0 = ((const uint4*)gA)[0], a1 = ((const uint4*)gA)[1];
        uint4 b0 = ((const uint4*)gB)[0], b1 = ((const uint4*)gB)[1];
        *(uint4*)(&sA[buf][lr][lc])     = a0;
        *(uint4*)(&sA[buf][lr][lc + 8]) = a1;
        *(uint4*)(&sB[buf][lr][lc])     = b0;
        *(uint4*)(&sB[buf][lr][lc + 8]) = b1;
#endif
        if (k0 + 32 < K) {               // pull chunk k0+32 toward the WGP
            __builtin_prefetch(gA + 32, 0, 3);
            __builtin_prefetch(gB + 32, 0, 3);
        }
    };

    issue_copy(0, 0);

    for (int c = 0; c < NC; ++c) {
        const int cur = c & 1;
        if (c + 1 < NC) issue_copy((c + 1) * 32, (c + 1) & 1);
#ifdef USE_ASYNC_LDS
        // my wave issued 4 async b128 for the next chunk; async ops complete
        // in order, so <=4 outstanding means the current chunk has landed.
        if (c + 1 < NC) WAIT_ASYNC(4); else WAIT_ASYNC(0);
#endif
        __syncthreads();                 // all waves' copies for chunk c done

        BF16x16 afrag[4], bfrag[2];
#pragma unroll
        for (int f = 0; f < 4; ++f) {
            const __bf16* p = &sA[cur][wm * 64 + f * 16 + mrow][0];
            afrag[f].h[0] = *(const v8bf*)(p + koffA);        // K koff..koff+7
            afrag[f].h[1] = *(const v8bf*)(p + koffA + 16);   // K koff+16..koff+23
        }
#pragma unroll
        for (int g = 0; g < 2; ++g) {
            const __bf16* p = &sB[cur][wn * 32 + g * 16 + mrow][0];
            bfrag[g].h[0] = *(const v8bf*)(p + koffB);        // K koffB..+7
            bfrag[g].h[1] = *(const v8bf*)(p + koffB + 16);   // K koffB+16..+23
        }
#pragma unroll
        for (int f = 0; f < 4; ++f)
#pragma unroll
            for (int g = 0; g < 2; ++g)
                acc[f][g] = __builtin_amdgcn_wmma_f32_16x16x32_bf16(
                    false, afrag[f].v, false, bfrag[g].v,
                    (short)0, acc[f][g], false, false);

        __syncthreads();                 // safe to overwrite buf cur at c+2
    }

    // C/D layout: v8f elem r -> M = r + hi*8 ; N = lane&15
#pragma unroll
    for (int g = 0; g < 2; ++g) {
        const int n  = n0 + wn * 32 + g * 16 + mrow;
        const float bv = bias[n];
#pragma unroll
        for (int f = 0; f < 4; ++f) {
            const int mb = m0 + wm * 64 + f * 16 + hi * 8;
#pragma unroll
            for (int r = 0; r < 8; ++r)
                Y[(size_t)(mb + r) * N + n] = acc[f][g][r] + bv;
        }
    }
}

// ---------------------------------------------------------------------------
extern "C" void kernel_launch(void* const* d_in, const int* in_sizes, int n_in,
                              void* d_out, int out_size, void* d_ws, size_t ws_size,
                              hipStream_t stream) {
    const float* x     = (const float*)d_in[0];
    const float* core0 = (const float*)d_in[1];
    const float* core1 = (const float*)d_in[2];
    const float* core2 = (const float*)d_in[3];
    const float* bias  = (const float*)d_in[4];
    float* Y = (float*)d_out;

    char* ws = (char*)d_ws;
    __bf16* Xb  = (__bf16*)(ws);
    __bf16* Wb  = (__bf16*)(ws + (size_t)8  * 1024 * 1024);
    float*  A01 = (float*) (ws + (size_t)40 * 1024 * 1024);

    tt_cast_x   <<<(BATCH * INF / 4) / 256, 256, 0, stream>>>(x, Xb);
    tt_combine01<<<(256 * 256 * 32) / 256,  256, 0, stream>>>(core0, core1, A01);
    tt_expand_w <<<(OUTF * INF) / 256,      256, 0, stream>>>(A01, core2, Wb);

    dim3 grid(OUTF / 128, BATCH / 128);     // 32 x 8 blocks
    tt_gemm_bias<<<grid, 256, 0, stream>>>(Xb, Wb, bias, Y);
}
